// localizer_26860725469574
// MI455X (gfx1250) — matrix operand order repaired and checked by reading, not compiled, not run on hardware
//
#include <hip/hip_runtime.h>
#include <hip/hip_bf16.h>

typedef __attribute__((ext_vector_type(16))) _Float16 v16h;
typedef __attribute__((ext_vector_type(8)))  _Float16 v8h;
typedef __attribute__((ext_vector_type(4)))  _Float16 v4h;
typedef __attribute__((ext_vector_type(2)))  _Float16 v2h;
typedef __attribute__((ext_vector_type(8)))  float    v8f;

#define NP   112   // 100 padded to 7*16
#define NLAY 5
#define DIM  64

static __device__ __forceinline__ v8h zero_v8h() {
    v8h z;
#pragma unroll
    for (int i = 0; i < 8; ++i) z[i] = (_Float16)0.0f;
    return z;
}

static __device__ __forceinline__ v16h cat16(v8h lo, v8h hi) {
    return __builtin_shufflevector(lo, hi, 0,1,2,3,4,5,6,7,8,9,10,11,12,13,14,15);
}

// One dense layer: act[16][NP] (f16, wave-private LDS) -> act, through
// W (LDS, [NP rows = out features][Ks cols = in features], f16, zero padded)
// bias (LDS f32 [NP], zero padded). fullK full 32-wide k-steps, optional
// 16-wide tail step (K = 96..111 real, 112..127 zeroed in registers).
static __device__ __forceinline__ void mlp_layer(
    const _Float16* __restrict__ W,
    const float*    __restrict__ bias,
    _Float16*       __restrict__ act,
    const int Ks, const int fullK, const bool tail, const int lane)
{
    const int lr = lane & 15;   // sample row (B/N index) & weight row (A/M index)
    const int hi = lane >> 4;

    v8f acc[7];
#pragma unroll
    for (int mt = 0; mt < 7; ++mt) {
        const float* bp = bias + mt * 16 + hi * 8;
        float4 b0 = *(const float4*)bp;
        float4 b1 = *(const float4*)(bp + 4);
        acc[mt] = (v8f){b0.x, b0.y, b0.z, b0.w, b1.x, b1.y, b1.z, b1.w};
    }

    for (int kt = 0; kt < fullK; ++kt) {
        const int kb = kt * 32;
        // B fragment: X^T, lane holds sample lr, K = kb + hi*8 + {0..7, 16..23}
        const _Float16* bp = act + lr * NP + kb + hi * 8;
        v16h B = cat16(*(const v8h*)bp, *(const v8h*)(bp + 16));
#pragma unroll
        for (int mt = 0; mt < 7; ++mt) {
            // A fragment: W^T row m = mt*16+lr; K = kb + hi*8 + {0..7, 16..23}
            const _Float16* ap = W + (mt * 16 + lr) * Ks + kb + hi * 8;
            v16h A = cat16(*(const v8h*)ap, *(const v8h*)(ap + 16));
            acc[mt] = __builtin_amdgcn_wmma_f32_16x16x32_f16(
                false, A, false, B, (short)0, acc[mt], false, false);
        }
    }

    if (tail) {  // kb = 96: K 96..111 valid, 112..127 forced zero
        const int kb = 96;
        v8h z = zero_v8h();
        v8h b0 = z, b1 = z;
        b0 = *(const v8h*)(act + lr * NP + kb + hi * 8);
        v16h B = cat16(b0, b1);
#pragma unroll
        for (int mt = 0; mt < 7; ++mt) {
            const _Float16* ap = W + (mt * 16 + lr) * Ks + kb + hi * 8;
            v16h A = cat16(*(const v8h*)ap, z);   // K>=112 contribution = 0
            acc[mt] = __builtin_amdgcn_wmma_f32_16x16x32_f16(
                false, A, false, B, (short)0, acc[mt], false, false);
        }
    }

    // ReLU -> f16, write back (all reads above are complete; in-wave LDS order)
#pragma unroll
    for (int mt = 0; mt < 7; ++mt) {
#pragma unroll
        for (int p = 0; p < 4; ++p) {
            float lo = fmaxf(acc[mt][2 * p + 0], 0.0f);
            float hh = fmaxf(acc[mt][2 * p + 1], 0.0f);
            v2h pk = { (_Float16)lo, (_Float16)hh };
            *(v2h*)(act + lr * NP + mt * 16 + hi * 8 + 2 * p) = pk;
        }
    }
}

__global__ __launch_bounds__(128)
void mlp_wmma_kernel(const float* __restrict__ x,
                     const float* __restrict__ Win,
                     const float* __restrict__ bin,
                     const float* __restrict__ Whid,
                     const float* __restrict__ bhid,
                     const float* __restrict__ Wout,
                     const float* __restrict__ bout,
                     float* __restrict__ out,
                     int rows)
{
    __shared__ _Float16 sWin[NP * DIM];        // [out n][in k], zero padded
    __shared__ _Float16 sWh[NLAY][NP * NP];    // [out n][in k], zero padded
    __shared__ float    sWout[NP];
    __shared__ float    sBin[NP];
    __shared__ float    sBh[NLAY][NP];
    __shared__ float    sBout;
    __shared__ _Float16 sAct[4][16 * NP];      // per-wave activation rows

    const int tid = threadIdx.x;

    // ---- one-time weight staging (transpose + pad + f16) ----
    for (int idx = tid; idx < NP * DIM; idx += 128) {
        int n = idx >> 6, k = idx & 63;
        sWin[idx] = (_Float16)((n < 100) ? Win[k * 100 + n] : 0.0f);
    }
    for (int l = 0; l < NLAY; ++l)
        for (int idx = tid; idx < NP * NP; idx += 128) {
            int n = idx / NP, k = idx - n * NP;
            sWh[l][idx] = (_Float16)((n < 100 && k < 100)
                                     ? Whid[l * 10000 + k * 100 + n] : 0.0f);
        }
    for (int idx = tid; idx < NP; idx += 128) {
        sWout[idx] = (idx < 100) ? Wout[idx] : 0.0f;
        sBin[idx]  = (idx < 100) ? bin[idx]  : 0.0f;
        for (int l = 0; l < NLAY; ++l)
            sBh[l][idx] = (idx < 100) ? bhid[l * 100 + idx] : 0.0f;
    }
    if (tid == 0) sBout = bout[0];
    __syncthreads();

    const int lane = tid & 31;
    const int wid  = tid >> 5;
    const int lr   = lane & 15;
    const int hi   = lane >> 4;
    _Float16* act  = sAct[wid];
    const float bO = sBout;
    const int nTiles = rows >> 6;   // 64 rows per block tile

    for (int tile = blockIdx.x; tile < nTiles; tile += gridDim.x) {
        const int rowBase = tile * 64 + wid * 16;

        // ---- stage 16 input rows (contiguous 4KB) as f16 into act ----
        const float* xb = x + (size_t)rowBase * DIM;
#pragma unroll
        for (int i = 0; i < 8; ++i) {
            int flat = lane * 4 + i * 128;       // float index in 16x64 tile
            float4 v = *(const float4*)(xb + flat);
            int r = flat >> 6, c = flat & 63;
            v4h h = { (_Float16)v.x, (_Float16)v.y, (_Float16)v.z, (_Float16)v.w };
            *(v4h*)(act + r * NP + c) = h;
        }

        // ---- 64 -> 100 ----
        mlp_layer(sWin, sBin, act, DIM, 2, false, lane);
        // ---- 5 x (100 -> 100) ----
        for (int l = 0; l < NLAY; ++l)
            mlp_layer(sWh[l], sBh[l], act, NP, 3, true, lane);

        // ---- 100 -> 1, abs ---- (cols >= 100 are exact zeros)
        float part = 0.0f;
        const _Float16* arow = act + lr * NP;
#pragma unroll
        for (int j = 0; j < 28; ++j) {
            int c = hi * 56 + 2 * j;
            v2h p = *(const v2h*)(arow + c);
            float2 w = *(const float2*)(sWout + c);
            part = fmaf((float)p[0], w.x, part);
            part = fmaf((float)p[1], w.y, part);
        }
        float tot = part + __shfl_xor(part, 16, 32);
        if (hi == 0)
            out[rowBase + lr] = fabsf(tot + bO);
    }
}

extern "C" void kernel_launch(void* const* d_in, const int* in_sizes, int n_in,
                              void* d_out, int out_size, void* d_ws, size_t ws_size,
                              hipStream_t stream) {
    const float* x    = (const float*)d_in[0];
    const float* Win  = (const float*)d_in[1];
    const float* bin  = (const float*)d_in[2];
    const float* Whid = (const float*)d_in[3];
    const float* bhid = (const float*)d_in[4];
    const float* Wout = (const float*)d_in[5];
    const float* bout = (const float*)d_in[6];
    float* out = (float*)d_out;

    const int rows = in_sizes[0] / DIM;
    int nTiles = rows >> 6;
    int grid = nTiles < 2048 ? (nTiles > 0 ? nTiles : 1) : 2048;

    mlp_wmma_kernel<<<grid, 128, 0, stream>>>(x, Win, bin, Whid, bhid,
                                              Wout, bout, out, rows);
}
